// DualModalAttention_10746008175332
// MI455X (gfx1250) — matrix operand order repaired and checked by reading, not compiled, not run on hardware
//
#include <hip/hip_runtime.h>
#include <math.h>

#define B_   8
#define C_   192
#define C3_  576
#define H_   128
#define W_   128
#define HW_  16384
#define NH_  8
#define CH_  24

typedef __attribute__((ext_vector_type(2)))  __bf16 v2bf;
typedef __attribute__((ext_vector_type(8)))  __bf16 v8bf;
typedef __attribute__((ext_vector_type(16))) __bf16 v16bf;
typedef __attribute__((ext_vector_type(8)))  float  v8f;

// ---------------------------------------------------------------------------
// GEMM: Y[b,O,HW] = W[O,K=192] * X[b,192,HW], fp32 in/out, bf16 WMMA core.
// Block = 256 threads = 8 wave32. Tile M=64, N=128, Kstep=32, K unrolled.
// LDS layouts are pre-swizzled so each lane's 16 fragment elements are a
// contiguous 32B run -> fragment build = 2x ds_load_b128, no packing VALU.
//   A: row-major, natural K order
//   B: transposed [n][k], natural K order
// Double-buffered LDS; one barrier per k-step.
// ---------------------------------------------------------------------------
#define TM 64
#define TN 128
#define KS 32
#define K_GEMM 192
#define NKS (K_GEMM / KS)
#define AP 40   // padded row length (80B): 16B-aligned, distinct banks

__global__ __launch_bounds__(256) void gemm_bf16_wmma(
    const float* __restrict__ Wt,   // [O][192]
    const float* __restrict__ X,    // [B][192][HW]
    float* __restrict__ Y,          // [B][O][HW]
    int O)
{
  __shared__ __align__(16) __bf16 As[2][TM][AP];
  __shared__ __align__(16) __bf16 Bs[2][TN][AP];

  const int b    = blockIdx.z;
  const int n0   = blockIdx.x * TN;
  const int m0   = blockIdx.y * TM;
  const int tid  = threadIdx.x;
  const int lane = tid & 31;
  const int wave = tid >> 5;
  const int wm   = (wave & 3) * 16;   // wave M offset inside tile
  const int wn   = (wave >> 2) * 64;  // wave N offset inside tile
  const int half = lane >> 4;
  const int l15  = lane & 15;

  const float* Xb = X + (size_t)b * K_GEMM * HW_;
  float*       Yb = Y + (size_t)b * O * HW_;

  // A staging: 256 units, each = 1 row x 8 K (2x float4 -> 1x v8bf store)
  const int sm  = tid >> 2;
  const int sk8 = (tid & 3) * 8;
  const int sp8 = sk8;

  auto stageA = [&](int k0, int buf) {
    const float* s = &Wt[(size_t)(m0 + sm) * K_GEMM + k0 + sk8];
    float4 f0 = *(const float4*)s;
    float4 f1 = *(const float4*)(s + 4);
    v8bf p;
    p[0] = (__bf16)f0.x; p[1] = (__bf16)f0.y; p[2] = (__bf16)f0.z; p[3] = (__bf16)f0.w;
    p[4] = (__bf16)f1.x; p[5] = (__bf16)f1.y; p[6] = (__bf16)f1.z; p[7] = (__bf16)f1.w;
    *(v8bf*)&As[buf][sm][sp8] = p;
  };

  // B staging: 512 units, each = 2 K-rows x 4 N (2x float4 -> 4x v2bf stores)
  auto stageB = [&](int k0, int buf) {
    #pragma unroll
    for (int u = 0; u < 2; ++u) {
      int uu = tid + u * 256;
      int kp = (uu >> 5) * 2;
      int nc = (uu & 31) * 4;
      const float* s = &Xb[(size_t)(k0 + kp) * HW_ + n0 + nc];
      float4 a = *(const float4*)s;
      float4 c = *(const float4*)(s + HW_);
      v2bf t0; t0[0] = (__bf16)a.x; t0[1] = (__bf16)c.x; *(v2bf*)&Bs[buf][nc + 0][kp] = t0;
      v2bf t1; t1[0] = (__bf16)a.y; t1[1] = (__bf16)c.y; *(v2bf*)&Bs[buf][nc + 1][kp] = t1;
      v2bf t2; t2[0] = (__bf16)a.z; t2[1] = (__bf16)c.z; *(v2bf*)&Bs[buf][nc + 2][kp] = t2;
      v2bf t3; t3[0] = (__bf16)a.w; t3[1] = (__bf16)c.w; *(v2bf*)&Bs[buf][nc + 3][kp] = t3;
    }
  };

  v8f zero = {};
  v8f acc[4] = {zero, zero, zero, zero};

  stageA(0, 0);
  stageB(0, 0);
  __syncthreads();

  #pragma unroll
  for (int s = 0; s < NKS; ++s) {
    const int cur = s & 1;
    if (s + 1 < NKS) {
      if (s + 2 < NKS)
        __builtin_prefetch(&Xb[(size_t)((s + 2) * KS + (tid >> 3)) * HW_ + n0 + (tid & 7) * 16], 0, 1);
      stageA((s + 1) * KS, cur ^ 1);
      stageB((s + 1) * KS, cur ^ 1);
    }

    // A fragment: contiguous 32B at half*16 within this lane's row
    const __bf16* arow = &As[cur][wm + l15][half * 16];
    v8bf alo = *(const v8bf*)arow;
    v8bf ahi = *(const v8bf*)(arow + 8);
    v16bf af = __builtin_shufflevector(alo, ahi,
        0, 1, 2, 3, 4, 5, 6, 7, 8, 9, 10, 11, 12, 13, 14, 15);

    #pragma unroll
    for (int t = 0; t < 4; ++t) {
      const __bf16* brow = &Bs[cur][wn + t * 16 + l15][half * 16];
      v8bf blo = *(const v8bf*)brow;
      v8bf bhi = *(const v8bf*)(brow + 8);
      v16bf bfr = __builtin_shufflevector(blo, bhi,
          0, 1, 2, 3, 4, 5, 6, 7, 8, 9, 10, 11, 12, 13, 14, 15);
      acc[t] = __builtin_amdgcn_wmma_f32_16x16x32_bf16(
          false, af, false, bfr, (short)0, acc[t], false, false);
    }
    __syncthreads();
  }

  // C/D layout: VGPR r, M = r + 8*(lane>>4), N = lane&15
  const int mb = m0 + wm + half * 8;
  const int nb = n0 + wn + l15;
  #pragma unroll
  for (int t = 0; t < 4; ++t)
    #pragma unroll
    for (int r = 0; r < 8; ++r)
      Yb[(size_t)(mb + r) * HW_ + nb + t * 16] = acc[t][r];
}

// ---------------------------------------------------------------------------
// Depthwise 3x3 SAME conv, NCHW. One thread per output pixel.
// ---------------------------------------------------------------------------
__global__ __launch_bounds__(256) void dwconv3x3_k(
    const float* __restrict__ in, const float* __restrict__ wdw,
    float* __restrict__ out)
{
  size_t idx = (size_t)blockIdx.x * 256 + threadIdx.x;
  int hw = (int)(idx & (HW_ - 1));
  size_t bc = idx >> 14;
  int c = (int)(bc % C3_);
  int x = hw & (W_ - 1);
  int y = hw >> 7;
  const float* w9 = wdw + c * 9;
  const float* p  = in + (bc << 14);
  float s = 0.f;
  #pragma unroll
  for (int dy = -1; dy <= 1; ++dy) {
    int yy = y + dy;
    if (yy < 0 || yy >= H_) continue;
    #pragma unroll
    for (int dx = -1; dx <= 1; ++dx) {
      int xx = x + dx;
      if (xx < 0 || xx >= W_) continue;
      s += w9[(dy + 1) * 3 + (dx + 1)] * p[yy * W_ + xx];
    }
  }
  out[idx] = s;
}

// ---------------------------------------------------------------------------
// Inverse L2 norms of q/k rows of both modalities. invn layout [mod][b][384].
// ---------------------------------------------------------------------------
__global__ __launch_bounds__(256) void rownorm_k(
    const float* __restrict__ qkv1, const float* __restrict__ qkv2,
    float* __restrict__ invn)
{
  int row = blockIdx.x;                 // mod*B*384 + b*384 + ch
  int ch  = row % 384;
  int b   = (row / 384) % B_;
  int mod = row / (384 * B_);
  const float* p = (mod ? qkv2 : qkv1) + ((size_t)b * C3_ + ch) * HW_;
  float s = 0.f;
  for (int i = threadIdx.x; i < HW_; i += 256) { float v = p[i]; s += v * v; }
  __shared__ float red[256];
  red[threadIdx.x] = s;
  __syncthreads();
  for (int off = 128; off > 0; off >>= 1) {
    if (threadIdx.x < off) red[threadIdx.x] += red[threadIdx.x + off];
    __syncthreads();
  }
  if (threadIdx.x == 0) invn[row] = 1.f / fmaxf(sqrtf(red[0]), 1e-12f);
}

__global__ void zero_k(float* __restrict__ p, int n) {
  int i = blockIdx.x * 256 + threadIdx.x;
  if (i < n) p[i] = 0.f;
}

// ---------------------------------------------------------------------------
// Partial 24x24 Gram (raw q·k) per (type, b*head, n-chunk of 256), atomicAdd.
// type: 0=q1·k1  1=q1·k2  2=q2·k2  3=q2·k1
// ---------------------------------------------------------------------------
__global__ __launch_bounds__(256) void gram_k(
    const float* __restrict__ qkv1, const float* __restrict__ qkv2,
    float* __restrict__ gram)
{
  __shared__ float qs[CH_][256];
  __shared__ float ks[CH_][256];
  int chunk = blockIdx.x;
  int bh    = blockIdx.y;
  int type  = blockIdx.z;
  int h = bh % NH_, b = bh / NH_;
  const float* qsrc = (type < 2) ? qkv1 : qkv2;
  const float* ksrc = (type == 0) ? qkv1 : (type == 1) ? qkv2
                    : (type == 2) ? qkv2 : qkv1;
  size_t qoff = ((size_t)b * C3_ + h * CH_) * HW_ + (size_t)chunk * 256;
  size_t koff = ((size_t)b * C3_ + 192 + h * CH_) * HW_ + (size_t)chunk * 256;
  for (int i = threadIdx.x; i < CH_ * 256; i += 256) {
    int c = i >> 8, j = i & 255;
    qs[c][j] = qsrc[qoff + (size_t)c * HW_ + j];
    ks[c][j] = ksrc[koff + (size_t)c * HW_ + j];
  }
  __syncthreads();
  for (int e = threadIdx.x; e < CH_ * CH_; e += 256) {
    int c = e / CH_, d = e % CH_;
    float s = 0.f;
    #pragma unroll 8
    for (int j = 0; j < 256; ++j) s += qs[c][j] * ks[d][j];
    atomicAdd(&gram[(((size_t)type * (B_ * NH_) + bh) * CH_ + c) * CH_ + d], s);
  }
}

// ---------------------------------------------------------------------------
// Build 24x24 softmax attention matrices from Grams, inv-norms, temperature.
// Cross terms (types 1,3) are negated before softmax.
// ---------------------------------------------------------------------------
__global__ void softmax_k(const float* __restrict__ gram,
                          const float* __restrict__ invn,
                          const float* __restrict__ temp,
                          float* __restrict__ attn)
{
  int idx  = blockIdx.x;
  int bh   = idx % (B_ * NH_);
  int type = idx / (B_ * NH_);
  int h = bh % NH_, b = bh / NH_;
  int c = threadIdx.x;
  if (c >= CH_) return;
  int mq = (type < 2) ? 0 : 1;
  int mk = (type == 0 || type == 3) ? 0 : 1;
  float iq  = invn[((size_t)mq * B_ + b) * 384 + h * CH_ + c];
  float tp  = temp[h];
  float sgn = (type == 1 || type == 3) ? -1.f : 1.f;
  const float* g = gram + (((size_t)type * (B_ * NH_) + bh) * CH_ + c) * CH_;
  float row[CH_];
  float mx = -1e30f;
  for (int d = 0; d < CH_; ++d) {
    float ik = invn[((size_t)mk * B_ + b) * 384 + 192 + h * CH_ + d];
    float v = sgn * g[d] * iq * ik * tp;
    row[d] = v;
    mx = fmaxf(mx, v);
  }
  float sum = 0.f;
  for (int d = 0; d < CH_; ++d) { row[d] = __expf(row[d] - mx); sum += row[d]; }
  float inv = 1.f / sum;
  float* o = attn + (((size_t)type * (B_ * NH_) + bh) * CH_ + c) * CH_;
  for (int d = 0; d < CH_; ++d) o[d] = row[d] * inv;
}

// ---------------------------------------------------------------------------
// out = sigmoid(beta) * (sa @ v_s) + (ca @ v_c), per (b,head), 256-col tiles.
// ---------------------------------------------------------------------------
__global__ __launch_bounds__(256) void apply_k(
    const float* __restrict__ attn,
    const float* __restrict__ qkv_s, const float* __restrict__ qkv_c,
    const float* __restrict__ beta, float* __restrict__ out,
    int ts, int tc)
{
  __shared__ float sa[CH_][CH_], ca[CH_][CH_];
  __shared__ float vs[CH_][256], vc[CH_][256];
  int n0  = blockIdx.x * 256;
  int bh  = blockIdx.y;
  int h = bh % NH_, b = bh / NH_;
  int tid = threadIdx.x;
  for (int i = tid; i < CH_ * CH_; i += 256) {
    sa[i / CH_][i % CH_] = attn[((size_t)ts * (B_ * NH_) + bh) * CH_ * CH_ + i];
    ca[i / CH_][i % CH_] = attn[((size_t)tc * (B_ * NH_) + bh) * CH_ * CH_ + i];
  }
  for (int d = 0; d < CH_; ++d) {
    vs[d][tid] = qkv_s[((size_t)b * C3_ + 384 + h * CH_ + d) * HW_ + n0 + tid];
    vc[d][tid] = qkv_c[((size_t)b * C3_ + 384 + h * CH_ + d) * HW_ + n0 + tid];
  }
  __syncthreads();
  float sb = 1.f / (1.f + __expf(-beta[0]));
  for (int c = 0; c < CH_; ++c) {
    float a0 = 0.f, a1 = 0.f;
    #pragma unroll
    for (int d = 0; d < CH_; ++d) {
      a0 += sa[c][d] * vs[d][tid];
      a1 += ca[c][d] * vc[d][tid];
    }
    out[((size_t)b * C_ + h * CH_ + c) * HW_ + n0 + tid] = sb * a0 + a1;
  }
}

// ---------------------------------------------------------------------------
extern "C" void kernel_launch(void* const* d_in, const int* in_sizes, int n_in,
                              void* d_out, int out_size, void* d_ws, size_t ws_size,
                              hipStream_t stream) {
  const float* x1   = (const float*)d_in[0];
  const float* x2   = (const float*)d_in[1];
  const float* wq1  = (const float*)d_in[2];
  const float* wd1  = (const float*)d_in[3];
  const float* wq2  = (const float*)d_in[4];
  const float* wd2  = (const float*)d_in[5];
  const float* wpr  = (const float*)d_in[6];
  const float* temp = (const float*)d_in[7];
  const float* beta = (const float*)d_in[8];

  float* ws = (float*)d_ws;
  const size_t SZ3 = (size_t)B_ * C3_ * HW_;   // 75,497,472 floats
  const size_t SZ1 = (size_t)B_ * C_  * HW_;   // 25,165,824 floats
  float* t    = ws;
  float* qkv1 = ws + SZ3;
  float* qkv2 = ws + 2 * SZ3;
  float* e1   = ws + 3 * SZ3;
  float* e2   = e1 + SZ1;
  float* invn = e2 + SZ1;
  float* gram = invn + 2 * B_ * 384;
  float* attn = gram + 4 * B_ * NH_ * CH_ * CH_;

  dim3 blk(256);
  const int GRAMN = 4 * B_ * NH_ * CH_ * CH_;

  // modality 1: qkv = dwconv(conv1x1(x1))
  gemm_bf16_wmma<<<dim3(HW_ / TN, C3_ / TM, B_), blk, 0, stream>>>(wq1, x1, t, C3_);
  dwconv3x3_k<<<dim3((unsigned)(SZ3 / 256)), blk, 0, stream>>>(t, wd1, qkv1);
  // modality 2
  gemm_bf16_wmma<<<dim3(HW_ / TN, C3_ / TM, B_), blk, 0, stream>>>(wq2, x2, t, C3_);
  dwconv3x3_k<<<dim3((unsigned)(SZ3 / 256)), blk, 0, stream>>>(t, wd2, qkv2);

  // attention core
  zero_k<<<dim3((GRAMN + 255) / 256), blk, 0, stream>>>(gram, GRAMN);
  rownorm_k<<<dim3(2 * B_ * 384), blk, 0, stream>>>(qkv1, qkv2, invn);
  gram_k<<<dim3(HW_ / 256, B_ * NH_, 4), blk, 0, stream>>>(qkv1, qkv2, gram);
  softmax_k<<<dim3(4 * B_ * NH_), dim3(32), 0, stream>>>(gram, invn, temp, attn);
  apply_k<<<dim3(HW_ / 256, B_ * NH_), blk, 0, stream>>>(attn, qkv1, qkv2, beta, e1, 0, 1);
  apply_k<<<dim3(HW_ / 256, B_ * NH_), blk, 0, stream>>>(attn, qkv2, qkv1, beta, e2, 2, 3);

  // projection GEMMs straight into d_out (out1 then out2)
  float* out = (float*)d_out;
  gemm_bf16_wmma<<<dim3(HW_ / TN, C_ / TM, B_), blk, 0, stream>>>(wpr, e1, out, C_);
  gemm_bf16_wmma<<<dim3(HW_ / TN, C_ / TM, B_), blk, 0, stream>>>(wpr, e2, out + SZ1, C_);
}
